// DepthAugmentedBEVLifter_8143257993862
// MI455X (gfx1250) — matrix operand order repaired and checked by reading, not compiled, not run on hardware
//
#include <hip/hip_runtime.h>
#include <hip/hip_fp16.h>

// ---------------------------------------------------------------------------
// DepthAugmentedBEVLifter for MI455X (gfx1250, wave32, WMMA, TDM).
//
// d_in mapping (top-level dict insertion order, params flattened as a jax
// pytree => dict keys sorted alphabetically, bn dicts sorted b,g,m,v):
//  0 feat_stage1 (6,64,128,352)   1 feat_stage3 (6,256,32,88)
//  2 feat_stage5 (6,512,16,44)    3 intrinsics (6,3,3)   4 extrinsics (6,4,4)
//  5..8  main.bn {b,g,m,v}(128)   9  main.w (128,64,3,3)
// 10..13 skip_stage1.bn (32)     14  skip_stage1.w (32,64)
// 15..18 skip_stage3.bn (128)    19  skip_stage3.w (128,64)
// 20..23 stage1.dep_bn (48)      24  stage1.dep_w (48,64)
// 25..28 stage1.red_bn (64)      29  stage1.red_w (64,64)
// 30..33 stage3.dep_bn (48)      34  stage3.dep_w (48,256)
// 35..38 stage3.red_bn (64)      39  stage3.red_w (64,256)
// 40..43 stage5.dep_bn (48)      44  stage5.dep_w (48,512)
// 45..48 stage5.red_bn (64)      49  stage5.red_w (64,512)
//
// d_out = [ main (6,128,128,128) | skip1 (6,32,128,128) | skip3 (6,128,128,128) ]  f32
// d_ws  = [ 3 x f16 BEV accumulators (6,16384,64) chan-fastest (37.7 MB)
//         | f16 WMMA-A-fragment-packed weights (354 KB) ]
// ---------------------------------------------------------------------------

typedef __attribute__((ext_vector_type(16))) _Float16 v16h;
typedef __attribute__((ext_vector_type(8)))  float    v8f;
typedef __attribute__((ext_vector_type(4)))  unsigned u32x4;
typedef __attribute__((ext_vector_type(8)))  unsigned u32x8;

union F16Frag { v16h v; _Float16 h[16]; uint4 q[2]; };

#define BEV   128
#define NPIX  16384      // 128*128
#define DCH   48

// halves per packed fragment set: one 16x32 fragment = 32 lanes * 16 halves
#define FRAG_H 512

__device__ __forceinline__ unsigned lds_addr_of(const void* p) {
  // flat LDS aperture pointer: low 32 bits == LDS byte offset
  return (unsigned)(unsigned long long)p;
}

// Tensor-Data-Mover 2D tile load: tile_d1 rows x tile_d0 f32 elements,
// row stride (elements) row_stride, global -> LDS (rows packed).
// D# group0/group1 per CDNA5 ISA 8.3/8.4; 2-group (2D) form, VADDR2/3 NULL.
__device__ __forceinline__ void tdm_load_2d(unsigned lds_off, const float* gsrc,
                                            unsigned tile_d0, unsigned tile_d1,
                                            unsigned long long row_stride) {
  unsigned long long ga = (unsigned long long)gsrc;
  u32x4 g0;
  g0[0] = 0x1u;                                            // count=1, user D#
  g0[1] = lds_off;                                         // lds_addr (bytes)
  g0[2] = (unsigned)(ga & 0xffffffffull);                  // global_addr[31:0]
  g0[3] = (unsigned)((ga >> 32) & 0x01ffffffull)           // global_addr[56:32]
        | (2u << 30);                                      // type = 2 (image)
  u32x8 g1;
  g1[0] = (2u << 16);                                      // data_size = 4B
  g1[1] = (tile_d0 & 0xffffu) << 16;                       // tensor_dim0 lo16
  g1[2] = ((tile_d0 >> 16) & 0xffffu)                      // tensor_dim0 hi16
        | ((tile_d1 & 0xffffu) << 16);                     // tensor_dim1 lo16
  g1[3] = ((tile_d1 >> 16) & 0xffffu)                      // tensor_dim1 hi16
        | ((tile_d0 & 0xffffu) << 16);                     // tile_dim0
  g1[4] = (tile_d1 & 0xffffu);                             // tile_dim1 (dim2=0)
  g1[5] = (unsigned)(row_stride & 0xffffffffull);          // dim0_stride lo32
  g1[6] = (unsigned)((row_stride >> 32) & 0xffffull);      // dim0_stride hi16
  g1[7] = 0u;
  asm volatile("tensor_load_to_lds %0, %1" :: "s"(g0), "s"(g1) : "memory");
}

__device__ __forceinline__ void wait_tensor() {
  __builtin_amdgcn_s_wait_tensorcnt(0);
}

__device__ __forceinline__ v8f wmma_f16(v16h a, v16h b, v8f c) {
  return __builtin_amdgcn_wmma_f32_16x16x32_f16(false, a, false, b,
                                                (short)0, c, false, false);
}

__device__ __forceinline__ void load_frag(F16Frag& f, const _Float16* pk,
                                          int frag, int lane) {
  const uint4* q = (const uint4*)(pk + ((size_t)frag * 32 + lane) * 16);
  f.q[0] = q[0];
  f.q[1] = q[1];
}

// ---------------------------------------------------------------------------
// Kernel 0: zero the f16 BEV accumulators (grid-stride float4 stores).
// ---------------------------------------------------------------------------
__global__ void zero_ws_kernel(float4* __restrict__ p, long long n4) {
  long long stride = (long long)gridDim.x * blockDim.x;
  for (long long i = (long long)blockIdx.x * blockDim.x + threadIdx.x;
       i < n4; i += stride)
    p[i] = make_float4(0.f, 0.f, 0.f, 0.f);
}

// ---------------------------------------------------------------------------
// Weight pre-packing: f32 [M][K] row-major -> f16 WMMA 16x32 A-fragments.
// Fragment f = kc*(M/16) + mt ; lane data = 32 contiguous bytes.
// A-layout (contiguous K order): lane<16 holds K 0..15,
// lane>=16 holds K 16..31 (16 consecutive halves).
// ---------------------------------------------------------------------------
__global__ void pack_dense_kernel(const float* __restrict__ W, int M, int K,
                                  _Float16* __restrict__ out) {
  int t = blockIdx.x * 256 + threadIdx.x;            // one (frag, lane)
  int total = (M / 16) * (K / 32) * 32;
  if (t >= total) return;
  int l  = t & 31;
  int f  = t >> 5;
  int mt = f % (M / 16);
  int kc = f / (M / 16);
  int row = mt * 16 + (l & 15);
  int base0 = (l < 16) ? 0 : 16;
  const float* src = W + (size_t)row * K + kc * 32;
  _Float16* dst = out + (size_t)t * 16;
#pragma unroll
  for (int j = 0; j < 8; ++j) dst[j]     = (_Float16)src[base0 + j];
#pragma unroll
  for (int j = 0; j < 8; ++j) dst[8 + j] = (_Float16)src[base0 + 8 + j];
}

// main.w (128,64,3,3) OIHW -> fragments f = (tap*2 + kc)*8 + ot, tap=dy*3+dx
__global__ void pack_conv3_kernel(const float* __restrict__ W,
                                  _Float16* __restrict__ out) {
  int t = blockIdx.x * 256 + threadIdx.x;
  if (t >= 144 * 32) return;
  int l   = t & 31;
  int f   = t >> 5;
  int ot  = f & 7;
  int kc  = (f >> 3) & 1;
  int tap = f >> 4;
  int oc  = ot * 16 + (l & 15);
  int base0 = (l < 16) ? 0 : 16;
  _Float16* dst = out + (size_t)t * 16;
#pragma unroll
  for (int j = 0; j < 8; ++j)
    dst[j]     = (_Float16)W[((size_t)oc * 64 + kc * 32 + base0 + j) * 9 + tap];
#pragma unroll
  for (int j = 0; j < 8; ++j)
    dst[8 + j] = (_Float16)W[((size_t)oc * 64 + kc * 32 + base0 + 8 + j) * 9 + tap];
}

// ---------------------------------------------------------------------------
// Kernel 1: per-scale lift + splat.
// Block = 256 threads (8 waves) / 64 pixels of one image.
// GEMM [112 out-ch] x [C] x [64 px] via v_wmma_f32_16x16x32_f16; features
// staged f32 global->LDS by the TDM, weights from pre-packed f16 fragments.
// Splat: f16 pk atomics into chan-fastest BEV accumulator.
// ---------------------------------------------------------------------------
struct LiftArgs {
  const float* feat;                       // [B][C][HW] f32
  const _Float16* pk_red;                  // packed [64][C] frags
  const _Float16* pk_dep;                  // packed [48][C] frags
  const float *rg, *rb, *rm, *rv;          // red_bn g,b,m,v (64)
  const float *dg, *db, *dm, *dv;          // dep_bn g,b,m,v (48)
  const float* K;                          // [B][9]
  const float* T;                          // [B][16]
  _Float16* acc;                           // [B][NPIX][64] f16
  int HW, Wimg;
};

template <int C>
__global__ __launch_bounds__(256) void lift_splat_kernel(LiftArgs A) {
  __shared__ float s_feat[32 * 64];        // K-chunk staging (f32)  8 KB
  __shared__ float s_out[64 * 112];        // GEMM results [px][ch] 28 KB
  __shared__ float s_sc[112], s_bi[112];   // fused BN scale/bias
  __shared__ float s_wgt[64];
  __shared__ int   s_idx[64];

  const int tid   = threadIdx.x;
  const int tiles = A.HW / 64;
  const int b     = blockIdx.x / tiles;
  const int hw0   = (blockIdx.x % tiles) * 64;

  if (tid < 112) {
    float g, bb, m, vv;
    if (tid < 64) { g = A.rg[tid]; bb = A.rb[tid]; m = A.rm[tid]; vv = A.rv[tid]; }
    else { int d = tid - 64; g = A.dg[d]; bb = A.db[d]; m = A.dm[d]; vv = A.dv[d]; }
    float sc = g * rsqrtf(vv + 1e-5f);
    s_sc[tid] = sc;
    s_bi[tid] = bb - m * sc;
  }

  const int lane = tid & 31;
  const int wid  = tid >> 5;
  const int pt   = wid & 3;                                  // pixel tile
  const int oh   = __builtin_amdgcn_readfirstlane(wid >> 2); // out half
  const int nt   = oh ? 3 : 4;
  const int p    = pt * 16 + (lane & 15);                    // pixel in block
  const int kb   = (lane < 16) ? 0 : 16;                     // B K-half

  v8f acc[4] = {};

  const float* feat_b = A.feat + (size_t)b * C * A.HW + hw0;
  __builtin_prefetch(A.pk_red, 0, 3);

  for (int kc = 0; kc < C / 32; ++kc) {
    __syncthreads();                       // previous chunk fully consumed
    {
      int wu = __builtin_amdgcn_readfirstlane(wid);
      const float* src = feat_b + (size_t)(kc * 32 + wu * 4) * A.HW;
      tdm_load_2d(lds_addr_of(&s_feat[wu * 4 * 64]), src, 64u, 4u,
                  (unsigned long long)A.HW);
      wait_tensor();
    }
    __syncthreads();                       // chunk visible to all waves

    F16Frag bf;                            // B: 32x16 (K x px)
#pragma unroll
    for (int j = 0; j < 16; ++j)
      bf.h[j] = (_Float16)s_feat[(kb + j) * 64 + p];

#pragma unroll
    for (int t = 0; t < 4; ++t) {
      if (t < nt) {
        const int ot = oh * 4 + t;
        F16Frag af;
        if (ot < 4) load_frag(af, A.pk_red, kc * 4 + ot, lane);
        else        load_frag(af, A.pk_dep, kc * 3 + (ot - 4), lane);
        acc[t] = wmma_f16(af.v, bf.v, acc[t]);
      }
    }
  }

  // dump accumulators to LDS as [pixel][channel]
#pragma unroll
  for (int t = 0; t < 4; ++t) {
    if (t < nt) {
      const int ot = oh * 4 + t;
#pragma unroll
      for (int e = 0; e < 8; ++e) {
        int ch = ot * 16 + e + ((lane >= 16) ? 8 : 0);
        s_out[p * 112 + ch] = acc[t][e];
      }
    }
  }
  __syncthreads();

  // per-pixel geometry: BN'd depth logits -> softmax(x10) -> expected depth,
  // unproject (K^-1), transform (T), voxelize
  if (tid < 64) {
    const int pp = tid;
    const int n  = hw0 + pp;
    const float x = (float)(n % A.Wimg);
    const float y = (float)(n / A.Wimg);

    float mx = -3.0e38f;
    for (int d = 0; d < DCH; ++d) {
      float l = (s_out[pp * 112 + 64 + d] * s_sc[64 + d] + s_bi[64 + d]) * 10.f;
      mx = fmaxf(mx, l);
    }
    float se = 0.f, sw = 0.f;
    for (int d = 0; d < DCH; ++d) {
      float l = (s_out[pp * 112 + 64 + d] * s_sc[64 + d] + s_bi[64 + d]) * 10.f;
      float e = __expf(l - mx);
      se += e;
      sw += e * __expf((float)d * 0.0871137141f);   // bins: 60^(d/47)
    }
    float depth = sw / se;

    const float* Km = A.K + b * 9;
    float k00 = Km[0], k01 = Km[1], k02 = Km[2];
    float k10 = Km[3], k11 = Km[4], k12 = Km[5];
    float k20 = Km[6], k21 = Km[7], k22 = Km[8];
    float det = k00 * (k11 * k22 - k12 * k21)
              - k01 * (k10 * k22 - k12 * k20)
              + k02 * (k10 * k21 - k11 * k20);
    float id = 1.f / det;
    float r0 = ((k11*k22 - k12*k21) * x + (k02*k21 - k01*k22) * y + (k01*k12 - k02*k11)) * id;
    float r1 = ((k12*k20 - k10*k22) * x + (k00*k22 - k02*k20) * y + (k02*k10 - k00*k12)) * id;
    float r2 = ((k10*k21 - k11*k20) * x + (k01*k20 - k00*k21) * y + (k00*k11 - k01*k10)) * id;
    float c0 = depth * r0, c1 = depth * r1, c2 = depth * r2;

    const float* Tm = A.T + b * 16;
    float e0 = Tm[0]*c0 + Tm[1]*c1 + Tm[2]*c2  + Tm[3];
    float e1 = Tm[4]*c0 + Tm[5]*c1 + Tm[6]*c2  + Tm[7];
    float e2 = Tm[8]*c0 + Tm[9]*c1 + Tm[10]*c2 + Tm[11];

    float bx = floorf(e0 * 2.5f + 64.f);
    float by = floorf(e1 * 2.5f + 64.f);
    bool valid = (bx >= 0.f) && (bx < 128.f) && (by >= 0.f) && (by < 128.f)
              && (e2 >= -5.f) && (e2 <= 3.f);
    s_idx[pp] = valid ? ((int)by * BEV + (int)bx) : -1;
    s_wgt[pp] = __expf(-0.05f * fabsf(e2));
  }
  __syncthreads();

  // splat: 4 threads per pixel, 16 channels each, packed-f16 global atomics
  {
    const int pp = tid & 63;
    const int cq = tid >> 6;
    const int ix = s_idx[pp];
    if (ix >= 0) {
      const float wgt = s_wgt[pp];
      _Float16* accb = A.acc + ((size_t)b * NPIX + ix) * 64;
#pragma unroll
      for (int cc = 0; cc < 8; ++cc) {
        int c = cq * 16 + cc * 2;
        float v0 = fmaxf(s_out[pp * 112 + c]     * s_sc[c]     + s_bi[c],     0.f) * wgt;
        float v1 = fmaxf(s_out[pp * 112 + c + 1] * s_sc[c + 1] + s_bi[c + 1], 0.f) * wgt;
        atomicAdd((__half2*)(accb + c), __floats2half2_rn(v0, v1));
      }
    }
  }
}

// ---------------------------------------------------------------------------
// Kernel 2: BEV 1x1 conv (64 -> OC) + BN + ReLU via WMMA.
// Accumulator is f16 chan-fastest, so the B fragment is a direct 32B load.
// Block: 8 waves, each owns one 16-pixel tile and all OC/16 out tiles.
// ---------------------------------------------------------------------------
struct ConvArgs {
  const _Float16* in;                      // [B][NPIX][64] f16
  const _Float16* pk;                      // packed [OC][64] frags
  const float *g, *b, *m, *v;
  float* out;                              // [B][OC][NPIX] f32
};

template <int OC>
__global__ __launch_bounds__(256) void bev_conv1x1_kernel(ConvArgs A) {
  __shared__ float s_sc[OC], s_bi[OC];

  const int tid  = threadIdx.x;
  const int lane = tid & 31;
  const int wid  = tid >> 5;
  const int b    = blockIdx.x / (NPIX / 128);
  const int hw0  = (blockIdx.x % (NPIX / 128)) * 128;

  for (int c = tid; c < OC; c += 256) {
    float sc = A.g[c] * rsqrtf(A.v[c] + 1e-5f);
    s_sc[c] = sc;
    s_bi[c] = A.b[c] - A.m[c] * sc;
  }
  __syncthreads();

  const int p  = hw0 + wid * 16 + (lane & 15);
  const int kb = (lane < 16) ? 0 : 16;
  const _Float16* in_p = A.in + ((size_t)b * NPIX + p) * 64;
  __builtin_prefetch(A.pk, 0, 3);

  v8f acc[OC / 16] = {};

#pragma unroll
  for (int kc = 0; kc < 2; ++kc) {
    F16Frag bf;
    const uint4* q = (const uint4*)(in_p + kc * 32 + kb);
    bf.q[0] = q[0];
    bf.q[1] = q[1];
#pragma unroll
    for (int ot = 0; ot < OC / 16; ++ot) {
      F16Frag af;
      load_frag(af, A.pk, kc * (OC / 16) + ot, lane);
      acc[ot] = wmma_f16(af.v, bf.v, acc[ot]);
    }
  }

  float* out_b = A.out + (size_t)b * OC * NPIX + p;
#pragma unroll
  for (int ot = 0; ot < OC / 16; ++ot) {
#pragma unroll
    for (int e = 0; e < 8; ++e) {
      int oc = ot * 16 + e + ((lane >= 16) ? 8 : 0);
      out_b[(size_t)oc * NPIX] = fmaxf(acc[ot][e] * s_sc[oc] + s_bi[oc], 0.f);
    }
  }
}

// ---------------------------------------------------------------------------
// Kernel 3: BEV 3x3 conv (64 -> 128, SAME) + BN + ReLU via WMMA.
// Wave owns a 16-pixel row segment; 2 K-chunks x 9 taps, B fragments are
// per-lane bounds-checked 32B loads (zero padding), A from packed frags.
// ---------------------------------------------------------------------------
struct Conv3Args {
  const _Float16* in;                      // [B][NPIX][64] f16
  const _Float16* pk;                      // packed conv3 frags (144)
  const float *g, *b, *m, *v;
  float* out;                              // [B][128][NPIX] f32
};

__global__ __launch_bounds__(256) void bev_conv3x3_kernel(Conv3Args A) {
  __shared__ float s_sc[128], s_bi[128];

  const int tid  = threadIdx.x;
  const int lane = tid & 31;
  const int wid  = tid >> 5;
  const int b    = blockIdx.x / 128;       // grid = B * 128 rows
  const int gy   = blockIdx.x % 128;
  const int gx0  = wid * 16;               // 8 waves cover the 128-wide row

  for (int c = tid; c < 128; c += 256) {
    float sc = A.g[c] * rsqrtf(A.v[c] + 1e-5f);
    s_sc[c] = sc;
    s_bi[c] = A.b[c] - A.m[c] * sc;
  }
  __syncthreads();

  const int kb = (lane < 16) ? 0 : 16;
  const int lx = lane & 15;
  __builtin_prefetch(A.pk, 0, 3);

  v8f acc[8] = {};

#pragma unroll
  for (int kc = 0; kc < 2; ++kc) {
    for (int tap = 0; tap < 9; ++tap) {
      const int dy = tap / 3, dx = tap % 3;
      const int yy = gy + dy - 1;
      const int xx = gx0 + lx + dx - 1;
      F16Frag bf;
      if (yy >= 0 && yy < 128 && xx >= 0 && xx < 128) {
        const uint4* q =
            (const uint4*)(A.in + (((size_t)b * NPIX + yy * 128 + xx) * 64
                                   + kc * 32 + kb));
        bf.q[0] = q[0];
        bf.q[1] = q[1];
      } else {
        bf.q[0] = make_uint4(0u, 0u, 0u, 0u);
        bf.q[1] = make_uint4(0u, 0u, 0u, 0u);
      }
#pragma unroll
      for (int ot = 0; ot < 8; ++ot) {
        F16Frag af;
        load_frag(af, A.pk, (tap * 2 + kc) * 8 + ot, lane);
        acc[ot] = wmma_f16(af.v, bf.v, acc[ot]);
      }
    }
  }

  float* out_b = A.out + ((size_t)b * 128) * NPIX + gy * 128 + gx0 + lx;
#pragma unroll
  for (int ot = 0; ot < 8; ++ot) {
#pragma unroll
    for (int e = 0; e < 8; ++e) {
      int oc = ot * 16 + e + ((lane >= 16) ? 8 : 0);
      out_b[(size_t)oc * NPIX] = fmaxf(acc[ot][e] * s_sc[oc] + s_bi[oc], 0.f);
    }
  }
}

// ---------------------------------------------------------------------------
extern "C" void kernel_launch(void* const* d_in, const int* in_sizes, int n_in,
                              void* d_out, int out_size, void* d_ws, size_t ws_size,
                              hipStream_t stream) {
  const float* f1 = (const float*)d_in[0];
  const float* f3 = (const float*)d_in[1];
  const float* f5 = (const float*)d_in[2];
  const float* K  = (const float*)d_in[3];
  const float* T  = (const float*)d_in[4];

  const float* main_bn_b = (const float*)d_in[5];
  const float* main_bn_g = (const float*)d_in[6];
  const float* main_bn_m = (const float*)d_in[7];
  const float* main_bn_v = (const float*)d_in[8];
  const float* main_w    = (const float*)d_in[9];

  const float* s1_bn_b = (const float*)d_in[10];
  const float* s1_bn_g = (const float*)d_in[11];
  const float* s1_bn_m = (const float*)d_in[12];
  const float* s1_bn_v = (const float*)d_in[13];
  const float* s1_w    = (const float*)d_in[14];

  const float* s3_bn_b = (const float*)d_in[15];
  const float* s3_bn_g = (const float*)d_in[16];
  const float* s3_bn_m = (const float*)d_in[17];
  const float* s3_bn_v = (const float*)d_in[18];
  const float* s3_w    = (const float*)d_in[19];

  const float* st_dep_bn_b[3] = {(const float*)d_in[20], (const float*)d_in[30], (const float*)d_in[40]};
  const float* st_dep_bn_g[3] = {(const float*)d_in[21], (const float*)d_in[31], (const float*)d_in[41]};
  const float* st_dep_bn_m[3] = {(const float*)d_in[22], (const float*)d_in[32], (const float*)d_in[42]};
  const float* st_dep_bn_v[3] = {(const float*)d_in[23], (const float*)d_in[33], (const float*)d_in[43]};
  const float* st_dep_w[3]    = {(const float*)d_in[24], (const float*)d_in[34], (const float*)d_in[44]};
  const float* st_red_bn_b[3] = {(const float*)d_in[25], (const float*)d_in[35], (const float*)d_in[45]};
  const float* st_red_bn_g[3] = {(const float*)d_in[26], (const float*)d_in[36], (const float*)d_in[46]};
  const float* st_red_bn_m[3] = {(const float*)d_in[27], (const float*)d_in[37], (const float*)d_in[47]};
  const float* st_red_bn_v[3] = {(const float*)d_in[28], (const float*)d_in[38], (const float*)d_in[48]};
  const float* st_red_w[3]    = {(const float*)d_in[29], (const float*)d_in[39], (const float*)d_in[49]};

  float* out = (float*)d_out;

  // workspace layout (halves)
  const long long acc_elems = 6LL * NPIX * 64;       // per scale, f16
  _Float16* acc1 = (_Float16*)d_ws;
  _Float16* acc3 = acc1 + acc_elems;
  _Float16* acc5 = acc3 + acc_elems;
  _Float16* pk   = acc5 + acc_elems;                 // packed weights base
  // packed fragment counts: red: (64/16)*(C/32), dep: (48/16)*(C/32)
  _Float16* pk_red1 = pk;                            //  8 frags
  _Float16* pk_dep1 = pk_red1 + 8   * FRAG_H;        //  6
  _Float16* pk_red3 = pk_dep1 + 6   * FRAG_H;        // 32
  _Float16* pk_dep3 = pk_red3 + 32  * FRAG_H;        // 24
  _Float16* pk_red5 = pk_dep3 + 24  * FRAG_H;        // 64
  _Float16* pk_dep5 = pk_red5 + 64  * FRAG_H;        // 48
  _Float16* pk_s1   = pk_dep5 + 48  * FRAG_H;        //  4
  _Float16* pk_s3   = pk_s1   + 4   * FRAG_H;        // 16
  _Float16* pk_m3   = pk_s3   + 16  * FRAG_H;        // 144

  // 0) zero f16 accumulators: 3 * 6 * 16384 * 64 halves = 37.75 MB
  const long long n4 = (3 * acc_elems) * 2 / 16;     // float4 count
  zero_ws_kernel<<<2048, 256, 0, stream>>>((float4*)d_ws, n4);

  // 0b) pre-pack all weights into f16 WMMA A-fragment layout
  pack_dense_kernel<<<1, 256, 0, stream>>>(st_red_w[0], 64, 64,  pk_red1);
  pack_dense_kernel<<<1, 256, 0, stream>>>(st_dep_w[0], 48, 64,  pk_dep1);
  pack_dense_kernel<<<4, 256, 0, stream>>>(st_red_w[1], 64, 256, pk_red3);
  pack_dense_kernel<<<3, 256, 0, stream>>>(st_dep_w[1], 48, 256, pk_dep3);
  pack_dense_kernel<<<8, 256, 0, stream>>>(st_red_w[2], 64, 512, pk_red5);
  pack_dense_kernel<<<6, 256, 0, stream>>>(st_dep_w[2], 48, 512, pk_dep5);
  pack_dense_kernel<<<1, 256, 0, stream>>>(s1_w,        32, 64,  pk_s1);
  pack_dense_kernel<<<2, 256, 0, stream>>>(s3_w,        128, 64, pk_s3);
  pack_conv3_kernel<<<18, 256, 0, stream>>>(main_w, pk_m3);

  // 1) lift + splat per scale
  {
    LiftArgs L;
    L.feat = f1; L.pk_red = pk_red1; L.pk_dep = pk_dep1;
    L.rg = st_red_bn_g[0]; L.rb = st_red_bn_b[0]; L.rm = st_red_bn_m[0]; L.rv = st_red_bn_v[0];
    L.dg = st_dep_bn_g[0]; L.db = st_dep_bn_b[0]; L.dm = st_dep_bn_m[0]; L.dv = st_dep_bn_v[0];
    L.K = K; L.T = T; L.acc = acc1; L.HW = 128 * 352; L.Wimg = 352;
    lift_splat_kernel<64><<<6 * (128 * 352 / 64), 256, 0, stream>>>(L);
  }
  {
    LiftArgs L;
    L.feat = f3; L.pk_red = pk_red3; L.pk_dep = pk_dep3;
    L.rg = st_red_bn_g[1]; L.rb = st_red_bn_b[1]; L.rm = st_red_bn_m[1]; L.rv = st_red_bn_v[1];
    L.dg = st_dep_bn_g[1]; L.db = st_dep_bn_b[1]; L.dm = st_dep_bn_m[1]; L.dv = st_dep_bn_v[1];
    L.K = K; L.T = T; L.acc = acc3; L.HW = 32 * 88; L.Wimg = 88;
    lift_splat_kernel<256><<<6 * (32 * 88 / 64), 256, 0, stream>>>(L);
  }
  {
    LiftArgs L;
    L.feat = f5; L.pk_red = pk_red5; L.pk_dep = pk_dep5;
    L.rg = st_red_bn_g[2]; L.rb = st_red_bn_b[2]; L.rm = st_red_bn_m[2]; L.rv = st_red_bn_v[2];
    L.dg = st_dep_bn_g[2]; L.db = st_dep_bn_b[2]; L.dm = st_dep_bn_m[2]; L.dv = st_dep_bn_v[2];
    L.K = K; L.T = T; L.acc = acc5; L.HW = 16 * 44; L.Wimg = 44;
    lift_splat_kernel<512><<<6 * (16 * 44 / 64), 256, 0, stream>>>(L);
  }

  // 2) BEV heads; d_out = [main | skip1 | skip3]
  const long long main_sz  = 6LL * 128 * NPIX;
  const long long skip1_sz = 6LL * 32 * NPIX;
  {
    ConvArgs Cg;
    Cg.in = acc1; Cg.pk = pk_s1;
    Cg.g = s1_bn_g; Cg.b = s1_bn_b; Cg.m = s1_bn_m; Cg.v = s1_bn_v;
    Cg.out = out + main_sz;
    bev_conv1x1_kernel<32><<<6 * (NPIX / 128), 256, 0, stream>>>(Cg);
  }
  {
    ConvArgs Cg;
    Cg.in = acc3; Cg.pk = pk_s3;
    Cg.g = s3_bn_g; Cg.b = s3_bn_b; Cg.m = s3_bn_m; Cg.v = s3_bn_v;
    Cg.out = out + main_sz + skip1_sz;
    bev_conv1x1_kernel<128><<<6 * (NPIX / 128), 256, 0, stream>>>(Cg);
  }
  {
    Conv3Args Cg;
    Cg.in = acc5; Cg.pk = pk_m3;
    Cg.g = main_bn_g; Cg.b = main_bn_b; Cg.m = main_bn_m; Cg.v = main_bn_v;
    Cg.out = out;
    bev_conv3x3_kernel<<<6 * 128, 256, 0, stream>>>(Cg);
  }
}
